// VNLinearLeakyReLU_68083821576960
// MI455X (gfx1250) — matrix operand order repaired and checked by reading, not compiled, not run on hardware
//
#include <hip/hip_runtime.h>

typedef __attribute__((ext_vector_type(16))) __bf16 v16bf;
typedef __attribute__((ext_vector_type(8)))  __bf16 v8bf;
typedef __attribute__((ext_vector_type(8)))  float  v8f;

#define CIN   256
#define COUT  256
#define NPTS  8192
#define BATCH 8
#define NORM_COUNT 65536   // BATCH * NPTS
#define WFRAG 65536        // bf16 elements per (matrix, half): 16 otile * 8 kstep * 32 lane * 16

__device__ __forceinline__ void split_bf16(float v, __bf16& hi, __bf16& lo) {
    __bf16 h = (__bf16)v;
    hi = h;
    lo = (__bf16)(v - (float)h);
}

// ---------------------------------------------------------------- pass 0a
__global__ __launch_bounds__(256) void vn_zero_stats(float* __restrict__ stats) {
    int t = threadIdx.x;
    stats[t]       = 0.0f;
    stats[t + 256] = 0.0f;
    stats[t + 512] = 0.0f;
    stats[t + 768] = 0.0f;
}

// ---------------------------------------------------------------- pass 0b
// Pre-split both weight matrices into bf16 hi/lo A-fragments in the exact
// wave32 WMMA A layout (ISA 7.12.2), so the GEMM loop loads fragments with
// coalesced b128 loads and does zero conversion VALU for A.
// One thread per (otile, kstep, lane) -> 16 bf16 elements.
__global__ __launch_bounds__(256) void vn_split_w(
    const float* __restrict__ Wf, const float* __restrict__ Wd,
    __bf16* __restrict__ fh, __bf16* __restrict__ fl,
    __bf16* __restrict__ dh, __bf16* __restrict__ dl)
{
    unsigned t     = blockIdx.x * 256u + threadIdx.x;   // 0..4095
    unsigned lane  = t & 31;
    unsigned kstep = (t >> 5) & 7;
    unsigned otile = t >> 8;
    unsigned M  = otile * 16 + (lane & 15);
    unsigned kb = kstep * 32 + (lane >> 4) * 8;

    const float* rf = Wf + (size_t)M * CIN + kb;
    const float* rd = Wd + (size_t)M * CIN + kb;
    __bf16* ofh = fh + (size_t)t * 16;
    __bf16* ofl = fl + (size_t)t * 16;
    __bf16* odh = dh + (size_t)t * 16;
    __bf16* odl = dl + (size_t)t * 16;

    #pragma unroll
    for (int e = 0; e < 8; ++e) {
        __bf16 h, l;
        split_bf16(rf[e],      h, l); ofh[e]     = h; ofl[e]     = l;
        split_bf16(rf[e + 16], h, l); ofh[e + 8] = h; ofl[e + 8] = l;
        split_bf16(rd[e],      h, l); odh[e]     = h; odl[e]     = l;
        split_bf16(rd[e + 16], h, l); odh[e + 8] = h; odl[e + 8] = l;
    }
}

// ---------------------------------------------------------------- pass 1
// grid: (NPTS/32, COUT/64, BATCH), block 256 (8 waves).
// Wave w: o-subtile = w>>1 (16 rows), n-subtile = w&1 (16 cols).
__global__ __launch_bounds__(256) void vn_gemm_stats(
    const float* __restrict__ x,
    const v16bf* __restrict__ aFh, const v16bf* __restrict__ aFl,
    const v16bf* __restrict__ aDh, const v16bf* __restrict__ aDl,
    float* __restrict__ p_out, float* __restrict__ d_buf,
    float* __restrict__ s1, float* __restrict__ s2)
{
    // x tile staged as pre-split bf16 hi/lo planes, transposed to [n][c],
    // row stride 40 bf16 (80 B: 16B-aligned rows, conflict-free strides)
    __shared__ __align__(16) __bf16 xhi[3][32][40];
    __shared__ __align__(16) __bf16 xlo[3][32][40];

    const int tid  = threadIdx.x;
    const int lane = tid & 31;
    const int wave = tid >> 5;
    const int osub = wave >> 1;
    const int nsub = wave & 1;
    const int ln   = lane & 15;
    const int lh   = lane >> 4;

    const int b      = blockIdx.z;
    const int oTile  = blockIdx.y * 4 + osub;       // 16-row tile index (0..15)
    const int o_base = oTile * 16;
    const int n_wg   = blockIdx.x * 32;
    const int n_base = n_wg + nsub * 16;

    v8f accP[3] = {};
    v8f accD[3] = {};

    for (int k0 = 0; k0 < CIN; k0 += 32) {
        // prefetch hint for next K tile (CDNA5 global_prefetch)
        if (k0 + 32 < CIN) {
            int c = tid >> 5, n = tid & 31;
            __builtin_prefetch(
                &x[(((size_t)b * CIN + (k0 + 32 + c)) * 3 + 0) * NPTS + (n_wg + n)], 0, 3);
        }

        // cooperative stage + split: 3 comps x 32c x 32n, stored transposed [n][c]
        #pragma unroll
        for (int i = 0; i < 12; ++i) {
            int idx = i * 256 + tid;          // 0..3071
            int d   = idx >> 10;
            int rem = idx & 1023;
            int c   = rem >> 5;
            int n   = rem & 31;
            float v = x[(((size_t)b * CIN + (k0 + c)) * 3 + d) * NPTS + (n_wg + n)];
            __bf16 h, l;
            split_bf16(v, h, l);
            xhi[d][n][c] = h;
            xlo[d][n][c] = l;
        }
        __syncthreads();

        // A fragments: direct b128 loads from pre-split, pre-swizzled arrays
        const unsigned fi = (unsigned)(oTile * 8 + (k0 >> 5)) * 32 + lane;
        v16bf afh = aFh[fi];
        v16bf afl = aFl[fi];
        v16bf adh = aDh[fi];
        v16bf adl = aDl[fi];

        // per vector component: B fragment = 2x ds_load_b128 per half, 6 WMMAs
        #pragma unroll
        for (int d = 0; d < 3; ++d) {
            const __bf16* ph = &xhi[d][nsub * 16 + ln][lh * 8];
            const __bf16* pl = &xlo[d][nsub * 16 + ln][lh * 8];
            v8bf h0 = *(const v8bf*)ph;
            v8bf h1 = *(const v8bf*)(ph + 16);
            v8bf l0 = *(const v8bf*)pl;
            v8bf l1 = *(const v8bf*)(pl + 16);
            v16bf bh = __builtin_shufflevector(h0, h1, 0,1,2,3,4,5,6,7,8,9,10,11,12,13,14,15);
            v16bf bl = __builtin_shufflevector(l0, l1, 0,1,2,3,4,5,6,7,8,9,10,11,12,13,14,15);

            accP[d] = __builtin_amdgcn_wmma_f32_16x16x32_bf16(false, afl, false, bh, (short)0, accP[d], false, false);
            accP[d] = __builtin_amdgcn_wmma_f32_16x16x32_bf16(false, afh, false, bl, (short)0, accP[d], false, false);
            accP[d] = __builtin_amdgcn_wmma_f32_16x16x32_bf16(false, afh, false, bh, (short)0, accP[d], false, false);

            accD[d] = __builtin_amdgcn_wmma_f32_16x16x32_bf16(false, adl, false, bh, (short)0, accD[d], false, false);
            accD[d] = __builtin_amdgcn_wmma_f32_16x16x32_bf16(false, adh, false, bl, (short)0, accD[d], false, false);
            accD[d] = __builtin_amdgcn_wmma_f32_16x16x32_bf16(false, adh, false, bh, (short)0, accD[d], false, false);
        }
        __syncthreads();
    }

    // write p / d tiles and accumulate BN statistics.
    // C/D layout: reg r, lane L -> M = r + 8*(L>=16), N = L&15
    #pragma unroll
    for (int r = 0; r < 8; ++r) {
        float p0 = accP[0][r], p1 = accP[1][r], p2 = accP[2][r];
        float nrm = sqrtf(p0 * p0 + p1 * p1 + p2 * p2);
        float s = nrm, q = nrm * nrm;
        #pragma unroll
        for (int off = 8; off >= 1; off >>= 1) {   // reduce within each 16-lane half
            s += __shfl_xor(s, off, 32);
            q += __shfl_xor(q, off, 32);
        }
        int o = o_base + r + lh * 8;
        if (ln == 0) {
            atomicAdd(&s1[o], s);
            atomicAdd(&s2[o], q);
        }
        size_t base = ((size_t)b * COUT + o) * 3 * NPTS + (n_base + ln);
        p_out[base]            = p0;
        p_out[base + NPTS]     = p1;
        p_out[base + 2 * NPTS] = p2;
        d_buf[base]            = accD[0][r];
        d_buf[base + NPTS]     = accD[1][r];
        d_buf[base + 2 * NPTS] = accD[2][r];
    }
}

// ---------------------------------------------------------------- pass 2
__global__ __launch_bounds__(256) void vn_bn_finalize(
    const float* __restrict__ s1, const float* __restrict__ s2,
    const float* __restrict__ gamma, const float* __restrict__ beta,
    float* __restrict__ scale, float* __restrict__ shift)
{
    int o = threadIdx.x;
    float mean = s1[o] * (1.0f / NORM_COUNT);
    float var  = s2[o] * (1.0f / NORM_COUNT) - mean * mean;   // biased, torch BN
    float sc   = gamma[o] * rsqrtf(var + 1e-5f);
    scale[o] = sc;
    shift[o] = beta[o] - mean * sc;
}

// ---------------------------------------------------------------- pass 3
__global__ __launch_bounds__(256) void vn_finish(
    float* __restrict__ out, const float* __restrict__ d_buf,
    const float* __restrict__ scale, const float* __restrict__ shift)
{
    unsigned i = blockIdx.x * 256u + threadIdx.x;     // over BATCH*COUT*NPTS
    int n = i & (NPTS - 1);
    int o = (i >> 13) & 255;
    size_t base = (size_t)(i >> 13) * 3 * NPTS + n;

    float p0 = out[base], p1 = out[base + NPTS], p2 = out[base + 2 * NPTS];
    float d0 = d_buf[base], d1 = d_buf[base + NPTS], d2 = d_buf[base + 2 * NPTS];

    float nrm = sqrtf(p0 * p0 + p1 * p1 + p2 * p2);
    float f   = (scale[o] * nrm + shift[o]) / nrm;    // p/|p| * norm_bn (as reference)
    p0 *= f; p1 *= f; p2 *= f;

    float dot = p0 * d0 + p1 * d1 + p2 * d2;
    if (dot < 0.0f) {
        float dsq  = d0 * d0 + d1 * d1 + d2 * d2;
        float coef = 0.8f * dot / (dsq + 1e-6f);      // (1 - NEG_SLOPE)
        p0 -= coef * d0; p1 -= coef * d1; p2 -= coef * d2;
    }
    out[base]            = p0;
    out[base + NPTS]     = p1;
    out[base + 2 * NPTS] = p2;
}

// ----------------------------------------------------------------
extern "C" void kernel_launch(void* const* d_in, const int* in_sizes, int n_in,
                              void* d_out, int out_size, void* d_ws, size_t ws_size,
                              hipStream_t stream) {
    const float* x     = (const float*)d_in[0];
    const float* Wf    = (const float*)d_in[1];
    const float* Wd    = (const float*)d_in[2];
    const float* gamma = (const float*)d_in[3];
    const float* beta  = (const float*)d_in[4];
    float* out = (float*)d_out;

    const size_t elems = (size_t)BATCH * COUT * 3 * NPTS;    // 50,331,648
    float* d_buf = (float*)d_ws;                             // direction GEMM scratch
    float* stats = (float*)((char*)d_ws + elems * sizeof(float));
    float* s1 = stats, *s2 = stats + 256, *sc = stats + 512, *sh = stats + 768;

    __bf16* wfrag = (__bf16*)((char*)d_ws + elems * sizeof(float) + 4096); // 32B aligned
    __bf16* fh = wfrag;
    __bf16* fl = wfrag + WFRAG;
    __bf16* dh = wfrag + 2 * WFRAG;
    __bf16* dl = wfrag + 3 * WFRAG;

    vn_zero_stats<<<1, 256, 0, stream>>>(stats);
    vn_split_w<<<16, 256, 0, stream>>>(Wf, Wd, fh, fl, dh, dl);

    dim3 grid(NPTS / 32, COUT / 64, BATCH);
    vn_gemm_stats<<<grid, 256, 0, stream>>>(
        x, (const v16bf*)fh, (const v16bf*)fl, (const v16bf*)dh, (const v16bf*)dl,
        out, d_buf, s1, s2);

    vn_bn_finalize<<<1, 256, 0, stream>>>(s1, s2, gamma, beta, sc, sh);

    unsigned total = (unsigned)(BATCH * COUT * NPTS);        // 16,777,216
    vn_finish<<<total / 256, 256, 0, stream>>>(out, d_buf, sc, sh);
}
